// KSpaceEncoder_86586540687378
// MI455X (gfx1250) — matrix operand order repaired and checked by reading, not compiled, not run on hardware
//
#include <hip/hip_runtime.h>

typedef _Float16 v16h __attribute__((ext_vector_type(16)));
typedef _Float16 v8h  __attribute__((ext_vector_type(8)));
typedef float    v8f  __attribute__((ext_vector_type(8)));

#define OUTD 64
#define BN_EPS 1e-5f

__device__ __forceinline__ void fatomic_add(float* p, float v) {
  // native global_atomic_add_f32 (no CAS loop)
  unsafeAtomicAdd(p, v);
}

// ---------------- utility kernels ----------------

__global__ void k_fill(float* __restrict__ p, float v, int n) {
  int i = blockIdx.x * blockDim.x + threadIdx.x;
  if (i < n) p[i] = v;
}

__global__ void k_deg(const long long* __restrict__ dst, float* __restrict__ deg, int E) {
  int e = blockIdx.x * blockDim.x + threadIdx.x;
  if (e < E) fatomic_add(&deg[(int)dst[e]], 1.0f);
}

__global__ void k_rsqrt_inplace(float* __restrict__ p, int n) {
  int i = blockIdx.x * blockDim.x + threadIdx.x;
  if (i < n) p[i] = rsqrtf(p[i]);
}

__global__ void k_f32_to_f16(const float* __restrict__ x, _Float16* __restrict__ xh, int n) {
  int i = blockIdx.x * blockDim.x + threadIdx.x;
  if (i < n) xh[i] = (_Float16)x[i];
}

// Pre-swizzle W [Din x 64] f32 row-major into per-lane WMMA B fragments (f16).
// Flat layout: [((kb*4 + ntile)*32 + lane)*16 + i]
// B-fragment ISA layout: half i of lane l holds B[kb*32 + (i&7) + (l>=16)*8 + (i>=8)*16][ntile*16 + (l&15)]
__global__ void k_packW(const float* __restrict__ W, _Float16* __restrict__ wp, int Din) {
  int tid = blockIdx.x * blockDim.x + threadIdx.x;
  int total = Din * 64;
  if (tid >= total) return;
  int i    = tid & 15;
  int lane = (tid >> 4) & 31;
  int t    = (tid >> 9) & 3;
  int kb   = tid >> 11;
  int hi   = lane >> 4;
  int K    = kb * 32 + (i & 7) + hi * 8 + (i >> 3) * 16;
  int col  = t * 16 + (lane & 15);
  wp[tid] = (_Float16)W[K * OUTD + col];
}

// ---------------- WMMA GEMM: H[N x 64] = Xh[N x Din] @ W ----------------
// one wave per 16x16 output tile; grid is an exact multiple of full waves.
__global__ void __launch_bounds__(256) k_gemm_wmma(
    const _Float16* __restrict__ Xh, const _Float16* __restrict__ Wp,
    float* __restrict__ H, int Din) {
  int widx  = (int)((blockIdx.x * blockDim.x + threadIdx.x) >> 5);
  int lane  = threadIdx.x & 31;
  int mtile = widx >> 2;
  int nt    = widx & 3;
  int hi    = lane >> 4;
  int m     = mtile * 16 + (lane & 15);

  v8f acc = {};
  int nkb = Din >> 5;
  for (int kb = 0; kb < nkb; ++kb) {
    // A fragment: half i of lane l = A[m][kb*32 + i + (i>=8)*8 + hi*8]
    const _Float16* pa = Xh + (size_t)m * Din + kb * 32 + hi * 8;
    v8h a0 = *(const v8h*)(pa);        // K = kb*32 + hi*8 + 0..7
    v8h a1 = *(const v8h*)(pa + 16);   // K = kb*32 + 16 + hi*8 + 0..7
    // B fragment: 32 contiguous bytes per lane from packed buffer
    const _Float16* pb = Wp + (((size_t)(kb * 4 + nt) * 32 + lane) << 4);
    v8h b0 = *(const v8h*)(pb);
    v8h b1 = *(const v8h*)(pb + 8);
    v16h A, B;
#pragma unroll
    for (int i = 0; i < 8; ++i) {
      A[i] = a0[i]; A[i + 8] = a1[i];
      B[i] = b0[i]; B[i + 8] = b1[i];
    }
    acc = __builtin_amdgcn_wmma_f32_16x16x32_f16(
        /*neg_a=*/false, A, /*neg_b=*/false, B,
        /*c_mod=*/(short)0, acc, /*reuse_a=*/false, /*reuse_b=*/false);
  }
  // D layout: acc[r] = D[r + hi*8][lane&15]
  float* ph = H + (size_t)(mtile * 16 + hi * 8) * OUTD + nt * 16 + (lane & 15);
#pragma unroll
  for (int r = 0; r < 8; ++r) ph[(size_t)r * OUTD] = acc[r];
}

// ---------------- aggregation ----------------

// AGG[n][c] = H[n][c] * dinv[n]^2 + bias[c]   (self-loop term + bias)
__global__ void k_selfinit(const float* __restrict__ H, const float* __restrict__ dinv,
                           const float* __restrict__ bias, float* __restrict__ AGG, int n64) {
  int idx = blockIdx.x * blockDim.x + threadIdx.x;
  if (idx >= n64) return;
  int n = idx >> 6, c = idx & 63;
  float di = dinv[n];
  AGG[idx] = H[idx] * di * di + bias[c];
}

// 16 threads per edge, float4 per thread
__global__ void k_edges(const long long* __restrict__ src, const long long* __restrict__ dst,
                        const float* __restrict__ dinv, const float* __restrict__ H,
                        float* __restrict__ AGG, int E) {
  int tid = blockIdx.x * blockDim.x + threadIdx.x;
  if (tid >= E * 16) return;
  int e = tid >> 4, q = tid & 15;
  int s = (int)src[e], d = (int)dst[e];
  float w = dinv[s] * dinv[d];
  const float4 h4 = *(const float4*)(H + (size_t)s * OUTD + q * 4);
  float* pd = AGG + (size_t)d * OUTD + q * 4;
  fatomic_add(pd + 0, h4.x * w);
  fatomic_add(pd + 1, h4.y * w);
  fatomic_add(pd + 2, h4.z * w);
  fatomic_add(pd + 3, h4.w * w);
}

// ---------------- batchnorm ----------------

// stats[0..63] += column sums, stats[64..127] += column sums of squares
__global__ void k_bnstats(const float* __restrict__ AGG, float* __restrict__ stats, int N) {
  __shared__ float ss[256], s2[256];
  int col = threadIdx.x & 63, rg = threadIdx.x >> 6;
  int chunk = (N + gridDim.x - 1) / gridDim.x;
  int r0 = blockIdx.x * chunk;
  int r1 = r0 + chunk; if (r1 > N) r1 = N;
  float a = 0.f, b = 0.f;
  for (int n = r0 + rg; n < r1; n += 4) {
    float v = AGG[(size_t)n * OUTD + col];
    a += v; b += v * v;
  }
  ss[threadIdx.x] = a; s2[threadIdx.x] = b;
  __syncthreads();
  if (threadIdx.x < 64) {
    a = ss[threadIdx.x] + ss[threadIdx.x + 64] + ss[threadIdx.x + 128] + ss[threadIdx.x + 192];
    b = s2[threadIdx.x] + s2[threadIdx.x + 64] + s2[threadIdx.x + 128] + s2[threadIdx.x + 192];
    fatomic_add(&stats[threadIdx.x], a);
    fatomic_add(&stats[64 + threadIdx.x], b);
  }
}

// xn = relu((v-mean)*gamma*rsqrt(var+eps) + beta); write f16 (next GEMM A input),
// optionally also f32 in place (for the final pooling pass).
__global__ void k_bnapply(const float* __restrict__ AGG, const float* __restrict__ stats,
                          const float* __restrict__ g, const float* __restrict__ be,
                          _Float16* __restrict__ Xh, float* __restrict__ outF32,
                          float invN, int n64) {
  int idx = blockIdx.x * blockDim.x + threadIdx.x;
  if (idx >= n64) return;
  int n = idx >> 6, c = idx & 63;
  float mean  = stats[c] * invN;
  float var   = stats[64 + c] * invN - mean * mean;
  float scale = g[c] * rsqrtf(var + BN_EPS);
  float v = (AGG[idx] - mean) * scale + be[c];
  v = fmaxf(v, 0.0f);
  Xh[(size_t)n * OUTD + c] = (_Float16)v;
  if (outF32) outF32[idx] = v;
}

// ---------------- pooling ----------------

__global__ void k_pool(const float* __restrict__ Hf, const long long* __restrict__ batch,
                       float* __restrict__ pool, float* __restrict__ cnt, int N) {
  int tid = blockIdx.x * blockDim.x + threadIdx.x;
  if (tid >= N * 16) return;
  int n = tid >> 4, q = tid & 15;
  int gI = (int)batch[n];
  const float4 v = *(const float4*)(Hf + (size_t)n * OUTD + q * 4);
  float* pp = pool + (size_t)gI * OUTD + q * 4;
  fatomic_add(pp + 0, v.x);
  fatomic_add(pp + 1, v.y);
  fatomic_add(pp + 2, v.z);
  fatomic_add(pp + 3, v.w);
  if (q == 0) fatomic_add(&cnt[gI], 1.0f);
}

__global__ void k_finalize(const float* __restrict__ pool, const float* __restrict__ cnt,
                           float* __restrict__ out, int n) {
  int i = blockIdx.x * blockDim.x + threadIdx.x;
  if (i >= n) return;
  out[i] = pool[i] / fmaxf(cnt[i >> 6], 1.0f);
}

// ---------------- host launcher ----------------

extern "C" void kernel_launch(void* const* d_in, const int* in_sizes, int n_in,
                              void* d_out, int out_size, void* d_ws, size_t ws_size,
                              hipStream_t stream) {
  const float*     x     = (const float*)d_in[0];
  const long long* ei    = (const long long*)d_in[1];
  const long long* batch = (const long long*)d_in[3];
  const float* W[3]  = {(const float*)d_in[4], (const float*)d_in[8],  (const float*)d_in[12]};
  const float* b[3]  = {(const float*)d_in[5], (const float*)d_in[9],  (const float*)d_in[13]};
  const float* g[3]  = {(const float*)d_in[6], (const float*)d_in[10], (const float*)d_in[14]};
  const float* be[3] = {(const float*)d_in[7], (const float*)d_in[11], (const float*)d_in[15]};
  float* out = (float*)d_out;

  const int N = in_sizes[0] / 128;   // 100000
  const int E = in_sizes[1] / 2;     // 1600000
  const long long* src = ei;
  const long long* dst = ei + E;

  // workspace carve (256B aligned)
  char* wsp = (char*)d_ws;
  auto carve = [&](size_t bytes) -> void* {
    void* p = (void*)wsp;
    wsp += (bytes + 255) & ~(size_t)255;
    return p;
  };
  float*    dinv  = (float*)   carve((size_t)N * 4);
  float*    H     = (float*)   carve((size_t)N * OUTD * 4);
  float*    AGG   = (float*)   carve((size_t)N * OUTD * 4);
  _Float16* Xh    = (_Float16*)carve((size_t)N * 128 * 2);
  _Float16* Wp    = (_Float16*)carve((size_t)128 * 64 * 2);
  float*    stats = (float*)   carve(128 * 4);
  float*    pool  = (float*)   carve((256 * 64 + 256) * 4);
  float*    cnt   = pool + 256 * 64;

  auto blk = [](int n) { return (n + 255) / 256; };

  // degree -> dinv (graph is the same for all three layers)
  k_fill<<<blk(N), 256, 0, stream>>>(dinv, 1.0f, N);            // self-loop
  k_deg<<<blk(E), 256, 0, stream>>>(dst, dinv, E);
  k_rsqrt_inplace<<<blk(N), 256, 0, stream>>>(dinv, N);

  // layer-1 input in f16
  k_f32_to_f16<<<blk(N * 128), 256, 0, stream>>>(x, Xh, N * 128);

  const float invN = 1.0f / (float)N;
  const int mtiles = N / 16;                       // 6250
  const int gemm_blocks = (mtiles * 4 + 7) / 8;    // 8 waves/block

  for (int L = 0; L < 3; ++L) {
    const int Din = (L == 0) ? 128 : 64;
    k_packW<<<blk(Din * 64), 256, 0, stream>>>(W[L], Wp, Din);
    k_gemm_wmma<<<gemm_blocks, 256, 0, stream>>>(Xh, Wp, H, Din);
    k_selfinit<<<blk(N * OUTD), 256, 0, stream>>>(H, dinv, b[L], AGG, N * OUTD);
    k_edges<<<blk(E * 16), 256, 0, stream>>>(src, dst, dinv, H, AGG, E);
    k_fill<<<1, 256, 0, stream>>>(stats, 0.0f, 128);
    k_bnstats<<<256, 256, 0, stream>>>(AGG, stats, N);
    k_bnapply<<<blk(N * OUTD), 256, 0, stream>>>(AGG, stats, g[L], be[L], Xh,
                                                 (L == 2) ? AGG : (float*)nullptr,
                                                 invN, N * OUTD);
  }

  // global mean pool
  k_fill<<<blk(256 * 64 + 256), 256, 0, stream>>>(pool, 0.0f, 256 * 64 + 256);
  k_pool<<<blk(N * 16), 256, 0, stream>>>(AGG, batch, pool, cnt, N);
  k_finalize<<<blk(256 * 64), 256, 0, stream>>>(pool, cnt, out, 256 * 64);
}
